// StatementEncoder_53532472378048
// MI455X (gfx1250) — matrix operand order repaired and checked by reading, not compiled, not run on hardware
//
#include <hip/hip_runtime.h>

#define FDIM 128

typedef __attribute__((ext_vector_type(16))) _Float16 v16h;
typedef __attribute__((ext_vector_type(8)))  _Float16 v8h;
typedef __attribute__((ext_vector_type(8)))  float    v8f;
typedef __attribute__((ext_vector_type(4)))  float    v4f;

__device__ __forceinline__ void atomicAddF32(float* p, float v) {
  // lowers to global_atomic_add_f32 on gfx1250 (no CAS loop)
  __hip_atomic_fetch_add(p, v, __ATOMIC_RELAXED, __HIP_MEMORY_SCOPE_AGENT);
}

// ---------------- init: deg=1 (self loop), pooling buffers = 0 ----------------
__global__ void k_init(float* deg, float* gsum, unsigned* gmax, int n) {
  int i = blockIdx.x * blockDim.x + threadIdx.x;
  if (i < n) deg[i] = 1.0f;
  if (i < FDIM) { gsum[i] = 0.0f; gmax[i] = 0u; }
}

// ---------------- degree accumulation over dst ----------------
__global__ void k_degree(const long long* ei, float* deg, int E) {
  int e = blockIdx.x * blockDim.x + threadIdx.x;
  if (e >= E) return;
  int d = (int)ei[(size_t)E + e];
  atomicAddF32(&deg[d], 1.0f);
}

__global__ void k_rsqrt(float* d, int n) {
  int i = blockIdx.x * blockDim.x + threadIdx.x;
  if (i < n) d[i] = rsqrtf(d[i]);   // deg >= 1 always (self loop)
}

// ---------------- f32 -> f16 convert (grid-stride) ----------------
__global__ void k_cvt_f16(const float* __restrict__ x, _Float16* __restrict__ y, size_t n) {
  size_t stride = (size_t)gridDim.x * blockDim.x;
  for (size_t i = blockIdx.x * (size_t)blockDim.x + threadIdx.x; i < n; i += stride)
    y[i] = (_Float16)x[i];
}

// ---------------- W [128x128] -> Wt f16 [col][k] ----------------
__global__ void k_wt_f16(const float* __restrict__ W, _Float16* __restrict__ Wt) {
  int idx = blockIdx.x * blockDim.x + threadIdx.x;   // 16384 total
  if (idx >= FDIM * FDIM) return;
  int col = idx >> 7, k = idx & 127;
  Wt[idx] = (_Float16)W[k * FDIM + col];
}

// ---------------- WMMA GEMM: H[N,128] = Xh[N,128] @ W[128,128] ----------------
// one wave = one 16x16 tile; 8 waves/block = 16 rows x 128 cols
__global__ void k_gemm_wmma(const _Float16* __restrict__ Xh,
                            const _Float16* __restrict__ Wth,
                            float* __restrict__ H, int nrows) {
  const int lane = threadIdx.x & 31;
  const int wv   = threadIdx.x >> 5;        // col tile 0..7
  const int mrow = blockIdx.x * 16;
  const int col0 = wv * 16;
  const int r    = lane & 15;
  const int hi   = lane >> 4;

  int arow = mrow + r; if (arow >= nrows) arow = nrows - 1;   // clamp (safe: row-local)
  const _Float16* ap = Xh  + (size_t)arow * FDIM + hi * 8;    // A: K = hi*8 + {0..7, +16..}
  const _Float16* bp = Wth + (size_t)(col0 + r) * FDIM + hi * 8;  // B: K = hi*8 + {0..7, +16..}

  v8f c = {};
#pragma unroll
  for (int kc = 0; kc < 4; ++kc) {
    union { v16h v; v8h h[2]; } a, b;
    a.h[0] = *(const v8h*)(ap + kc * 32);
    a.h[1] = *(const v8h*)(ap + kc * 32 + 16);
    b.h[0] = *(const v8h*)(bp + kc * 32);
    b.h[1] = *(const v8h*)(bp + kc * 32 + 16);
    c = __builtin_amdgcn_wmma_f32_16x16x32_f16(false, a.v, false, b.v,
                                               (short)0, c, false, false);
  }

  // D layout: lane<16 -> rows mrow+0..7, lane>=16 -> rows mrow+8..15; col = col0 + (lane&15)
  const int n = col0 + (lane & 15);
  const int mbase = mrow + hi * 8;
  float* outp = H + (size_t)mbase * FDIM + n;
  if (mrow + 16 <= nrows) {
    // full tile (always the case when nrows % 16 == 0): straight-line stores
#pragma unroll
    for (int t = 0; t < 8; ++t)
      outp[(size_t)t * FDIM] = c[t];
  } else {
    // generic tail tile
#pragma unroll
    for (int t = 0; t < 8; ++t)
      if (mbase + t < nrows) outp[(size_t)t * FDIM] = c[t];
  }
}

// ---------------- zero f32 buffer ----------------
__global__ void k_zero(float* p, size_t n) {
  size_t stride = (size_t)gridDim.x * blockDim.x;
  for (size_t i = blockIdx.x * (size_t)blockDim.x + threadIdx.x; i < n; i += stride)
    p[i] = 0.0f;
}

// ---------------- edge scatter: acc[dst] += h[src] * dinv[src]*dinv[dst] ------
// one wave per edge, 4 features per lane
__global__ void k_edges(const long long* __restrict__ ei,
                        const float* __restrict__ dinv,
                        const float* __restrict__ h,
                        float* __restrict__ acc, int E) {
  size_t tid = blockIdx.x * (size_t)blockDim.x + threadIdx.x;
  size_t e = tid >> 5;
  if (e >= (size_t)E) return;
  int f4 = (int)(tid & 31) * 4;
  int s = (int)ei[e];
  int d = (int)ei[(size_t)E + e];
  float w = dinv[s] * dinv[d];
  v4f hv = *(const v4f*)(h + (size_t)s * FDIM + f4);
  float* apx = acc + (size_t)d * FDIM + f4;
  atomicAddF32(apx + 0, hv[0] * w);
  atomicAddF32(apx + 1, hv[1] * w);
  atomicAddF32(apx + 2, hv[2] * w);
  atomicAddF32(apx + 3, hv[3] * w);
}

// ---------------- epilogue: x = relu(acc + h*dinv^2 + b); keep f32 + f16 ------
__global__ void k_finalize(float* __restrict__ acc, const float* __restrict__ h,
                           const float* __restrict__ dinv, const float* __restrict__ bias,
                           _Float16* __restrict__ xh, size_t n) {
  size_t stride = (size_t)gridDim.x * blockDim.x;
  for (size_t idx = blockIdx.x * (size_t)blockDim.x + threadIdx.x; idx < n; idx += stride) {
    size_t i = idx >> 7;
    int f = (int)(idx & 127);
    float di = dinv[i];
    float v = acc[idx] + h[idx] * di * di + bias[f];
    v = fmaxf(v, 0.0f);
    acc[idx] = v;
    xh[idx]  = (_Float16)v;
  }
}

// ---------------- pooling: column sum + max (post-ReLU => >= 0) ----------------
__global__ void k_pool(const float* __restrict__ x, float* gsum, unsigned* gmax, int nrows) {
  const int f = threadIdx.x;                 // 128 threads
  int r0 = blockIdx.x * 256;
  int r1 = r0 + 256; if (r1 > nrows) r1 = nrows;
  float s = 0.0f, m = 0.0f;
  for (int rr = r0; rr < r1; ++rr) {
    float v = x[(size_t)rr * FDIM + f];
    s += v; m = fmaxf(m, v);
  }
  atomicAddF32(&gsum[f], s);
  atomicMax(&gmax[f], __float_as_uint(m));   // valid: values are non-negative
}

// ---------------- final FC: out[256] = [mean|max] @ fcW + fcb ----------------
__global__ void k_fc(const float* __restrict__ gsum, const unsigned* __restrict__ gmax,
                     const float* __restrict__ fcW, const float* __restrict__ fcb,
                     float* __restrict__ out, float invN) {
  __shared__ float p[2 * FDIM];
  int t = threadIdx.x;                       // 256 threads
  if (t < FDIM) {
    p[t] = gsum[t] * invN;
    p[FDIM + t] = __uint_as_float(gmax[t]);
  }
  __syncthreads();
  float a = fcb[t];
  for (int k = 0; k < 2 * FDIM; ++k)
    a += p[k] * fcW[k * 256 + t];
  out[t] = a;
}

extern "C" void kernel_launch(void* const* d_in, const int* in_sizes, int n_in,
                              void* d_out, int out_size, void* d_ws, size_t ws_size,
                              hipStream_t stream) {
  (void)n_in; (void)out_size; (void)ws_size;
  const float*     X   = (const float*)d_in[0];
  const long long* ei  = (const long long*)d_in[1];   // int64 in reference
  const float*     W1  = (const float*)d_in[2];
  const float*     b1  = (const float*)d_in[3];
  const float*     W2  = (const float*)d_in[4];
  const float*     b2  = (const float*)d_in[5];
  const float*     fcW = (const float*)d_in[6];
  const float*     fcb = (const float*)d_in[7];
  float* out = (float*)d_out;

  const int N  = in_sizes[0] / FDIM;
  const int E  = in_sizes[1] / 2;
  const size_t NF = (size_t)N * FDIM;

  // ---- workspace carve (16B aligned) ----
  char* base = (char*)d_ws;
  size_t off = 0;
  auto AL = [](size_t x) { return (x + 15) & ~(size_t)15; };
  float*    dinv = (float*)(base + off);    off = AL(off + (size_t)N * 4);
  float*    hbuf = (float*)(base + off);    off = AL(off + NF * 4);
  float*    acc  = (float*)(base + off);    off = AL(off + NF * 4);
  _Float16* xh   = (_Float16*)(base + off); off = AL(off + NF * 2);
  _Float16* wth  = (_Float16*)(base + off); off = AL(off + (size_t)FDIM * FDIM * 2);
  float*    gsum = (float*)(base + off);    off = AL(off + FDIM * 4);
  unsigned* gmax = (unsigned*)(base + off); off = AL(off + FDIM * 4);

  const int T = 256;
  const unsigned gN   = (unsigned)((N + T - 1) / T);
  const unsigned gE   = (unsigned)((E + T - 1) / T);
  const unsigned gNF  = (unsigned)((NF + T - 1) / T);
  const unsigned gEdg = (unsigned)(((size_t)E * 32 + T - 1) / T);
  const unsigned gGemm = (unsigned)((N + 15) / 16);
  const unsigned gW   = (FDIM * FDIM + T - 1) / T;
  const unsigned gPool = (unsigned)((N + 255) / 256);

  // norm: deg -> dinv
  k_init  <<<gN, T, 0, stream>>>(dinv, gsum, gmax, N);
  k_degree<<<gE, T, 0, stream>>>(ei, dinv, E);
  k_rsqrt <<<gN, T, 0, stream>>>(dinv, N);

  // f16 copy of node features
  k_cvt_f16<<<gNF, T, 0, stream>>>(X, xh, NF);

  // ---- layer 1 ----
  k_wt_f16   <<<gW, T, 0, stream>>>(W1, wth);
  k_gemm_wmma<<<gGemm, T, 0, stream>>>(xh, wth, hbuf, N);
  k_zero     <<<gNF, T, 0, stream>>>(acc, NF);
  k_edges    <<<gEdg, T, 0, stream>>>(ei, dinv, hbuf, acc, E);
  k_finalize <<<gNF, T, 0, stream>>>(acc, hbuf, dinv, b1, xh, NF);

  // ---- layer 2 ----
  k_wt_f16   <<<gW, T, 0, stream>>>(W2, wth);
  k_gemm_wmma<<<gGemm, T, 0, stream>>>(xh, wth, hbuf, N);
  k_zero     <<<gNF, T, 0, stream>>>(acc, NF);
  k_edges    <<<gEdg, T, 0, stream>>>(ei, dinv, hbuf, acc, E);
  k_finalize <<<gNF, T, 0, stream>>>(acc, hbuf, dinv, b2, xh, NF);

  // ---- pool + fc ----
  k_pool<<<gPool, FDIM, 0, stream>>>(acc, gsum, gmax, N);
  k_fc  <<<1, 256, 0, stream>>>(gsum, gmax, fcW, fcb, out, 1.0f / (float)N);
}
